// RNNWavefunction_71597104824779
// MI455X (gfx1250) — matrix operand order repaired and checked by reading, not compiled, not run on hardware
//
#include <hip/hip_runtime.h>
#include <hip/hip_bf16.h>
#include <stdint.h>

#define SORB  64
#define HDIM  512
#define BT    64            // batch rows per workgroup
#define HP    (HDIM + 8)    // padded LDS row stride (bank-conflict avoidance)
#define HBUF  (BT * HP)     // elements per hidden buffer
#define WROWS (3 * HDIM)    // 1536
#define WELEM (WROWS * HDIM)

typedef __attribute__((ext_vector_type(16))) __bf16 v16bf;
typedef __attribute__((ext_vector_type(8)))  float  v8f;

__device__ __forceinline__ float bf2f(unsigned short h) {
  return __uint_as_float(((unsigned int)h) << 16);
}
__device__ __forceinline__ unsigned short f2bf(float f) {
  unsigned int u = __float_as_uint(f);
  u += 0x7fffu + ((u >> 16) & 1u);      // round-to-nearest-even
  return (unsigned short)(u >> 16);
}
__device__ __forceinline__ float bflo(unsigned int w) { return __uint_as_float(w << 16); }
__device__ __forceinline__ float bfhi(unsigned int w) { return __uint_as_float(w & 0xffff0000u); }

__device__ __forceinline__ v16bf mkfrag(uint4 a, uint4 b) {
  union { uint4 q[2]; v16bf v; } u; u.q[0] = a; u.q[1] = b; return u.v;
}
// A fragment: 16x32 bf16, rows m0..m0+15 from LDS (row stride HP).
// lanes 0-15: K = k0 + 0..7 (q0), k0+16..23 (q1); lanes 16-31: +8.
__device__ __forceinline__ v16bf loadA_lds(const unsigned short* h, int m0, int k0,
                                           int l16, int half) {
  const uint4* q = (const uint4*)h;
  int base = (((m0 + l16) * HP) + k0 + half * 8) >> 3;   // uint4 index (8 bf16 each)
  return mkfrag(q[base], q[base + 2]);
}
// B fragment: 32x16 (KxN) bf16 from row-major weights (N = weight row, stride HDIM).
// lane = N column; lanes 0-15 hold K = k0+0..7, k0+16..23; lanes 16-31: +8.
__device__ __forceinline__ v16bf loadB_g(const unsigned short* w, int n0, int k0,
                                         int l16, int half) {
  const uint4* q = (const uint4*)w;
  int base = (((n0 + l16) * HDIM) + k0 + half * 8) >> 3;
  return mkfrag(q[base], q[base + 2]);
}
__device__ __forceinline__ v8f wmma_bf(v16bf a, v16bf b, v8f c) {
  return __builtin_amdgcn_wmma_f32_16x16x32_bf16(false, a, false, b, (short)0, c,
                                                 false, false);
}
__device__ __forceinline__ float sigf(float x) { return 1.0f / (1.0f + __expf(-x)); }

// ---- weight fp32 -> bf16 conversion into workspace ----
__global__ void __launch_bounds__(256) rnnwf_cvt(const float* __restrict__ a,
                                                 const float* __restrict__ b,
                                                 const float* __restrict__ c,
                                                 unsigned short* __restrict__ o) {
  int i = blockIdx.x * 256 + threadIdx.x;
  if (i < WELEM)               o[i] = f2bf(a[i]);
  else if (i < 2 * WELEM)      o[i] = f2bf(b[i - WELEM]);
  else if (i < 3 * WELEM)      o[i] = f2bf(c[i - 2 * WELEM]);
}

// ---- main persistent scan kernel ----
__global__ void __launch_bounds__(256, 1) rnnwf_main(
    const int* __restrict__ xg,
    const float* __restrict__ Wih0,
    const unsigned short* __restrict__ Whh0,
    const unsigned short* __restrict__ Wih1,
    const unsigned short* __restrict__ Whh1,
    const float* __restrict__ Wamp, const float* __restrict__ bamp,
    const float* __restrict__ Wph,  const float* __restrict__ bph,
    float* __restrict__ out) {
  extern __shared__ char smem[];
  unsigned short* hA   = (unsigned short*)smem;   // [2][HBUF] layer-0 hidden (bf16)
  unsigned short* hB   = hA + 2 * HBUF;           // [2][HBUF] layer-1 hidden (bf16)
  float* headsum = (float*)(smem + 4 * HBUF * 2); // 256 floats (64 rows x 4 heads)
  float* ampL  = headsum + 256;                   // 64
  float* phL   = ampL + BT;
  float* nupA  = phL + BT;
  float* ndnA  = nupA + BT;
  int*   bprev = (int*)(ndnA + BT);               // 64

  const int t    = threadIdx.x;
  const int w    = t >> 5;
  const int lane = t & 31;
  const int l16  = lane & 15;
  const int half = lane >> 4;
  const int m0    = (w & 3) * 16;       // M-tile of this wave
  const int jbase = (w >> 2) * 16;      // N-chunk range of this wave
  const int rowg0 = blockIdx.x * BT;

  { // init: zero hidden buffer 0, per-row state
    unsigned int* za = (unsigned int*)hA;
    unsigned int* zb = (unsigned int*)hB;
    for (int i = t; i < HBUF / 2; i += 256) { za[i] = 0u; zb[i] = 0u; }
    if (t < BT) { ampL[t] = 1.f; phL[t] = 0.f; nupA[t] = 0.f; ndnA[t] = 0.f; bprev[t] = -1; }
  }
  __syncthreads();

  int p = 0;
  #pragma unroll 1
  for (int i = 0; i < SORB; ++i) {
    const unsigned short* h0old = hA + p * HBUF;
    unsigned short*       h0new = hA + (p ^ 1) * HBUF;
    const unsigned short* h1old = hB + p * HBUF;
    unsigned short*       h1new = hB + (p ^ 1) * HBUF;

    // ---------- layer 0: gh = h0 @ Whh0^T, gi = column select of Wih0 ----------
    #pragma unroll 1
    for (int tt = 0; tt < 16; ++tt) {
      int col0 = (jbase + tt) * 16;
      v8f aR = {0.f,0.f,0.f,0.f,0.f,0.f,0.f,0.f};
      v8f aZ = aR, aN = aR;
      #pragma unroll
      for (int ks = 0; ks < 16; ++ks) {
        int k0 = ks * 32;
        v16bf af = loadA_lds(h0old, m0, k0, l16, half);
        v16bf br = loadB_g(Whh0, col0,            k0, l16, half);
        v16bf bz = loadB_g(Whh0, HDIM + col0,     k0, l16, half);
        v16bf bn = loadB_g(Whh0, 2 * HDIM + col0, k0, l16, half);
        aR = wmma_bf(af, br, aR);
        aZ = wmma_bf(af, bz, aZ);
        aN = wmma_bf(af, bn, aN);
      }
      int col = col0 + l16;
      #pragma unroll
      for (int e = 0; e < 8; ++e) {
        int row = m0 + half * 8 + e;
        int bp  = bprev[row];
        int bi  = bp < 0 ? 0 : bp;
        float on = bp < 0 ? 0.f : 1.f;
        float gir = on * Wih0[(col) * 2 + bi];
        float giz = on * Wih0[(HDIM + col) * 2 + bi];
        float gin = on * Wih0[(2 * HDIM + col) * 2 + bi];
        float r = sigf(aR[e] + gir);
        float z = sigf(aZ[e] + giz);
        float n = tanhf(gin + r * aN[e]);
        float ho = bf2f(h0old[row * HP + col]);
        h0new[row * HP + col] = f2bf((1.f - z) * n + z * ho);
      }
    }
    __syncthreads();

    // ---------- layer 1: r,z over concat K=1024; n-gate split gi/gh ----------
    #pragma unroll 1
    for (int tt = 0; tt < 16; ++tt) {
      int col0 = (jbase + tt) * 16;
      v8f aR = {0.f,0.f,0.f,0.f,0.f,0.f,0.f,0.f};
      v8f aZ = aR, aGi = aR, aGh = aR;
      #pragma unroll
      for (int ks = 0; ks < 32; ++ks) {
        const bool second = (ks >= 16);           // compile-time under unroll
        int kk = (ks & 15) * 32;
        const unsigned short* Asrc = second ? h1old : h0new;
        const unsigned short* Wsrc = second ? Whh1  : Wih1;
        v16bf af = loadA_lds(Asrc, m0, kk, l16, half);
        v16bf br = loadB_g(Wsrc, col0,            kk, l16, half);
        v16bf bz = loadB_g(Wsrc, HDIM + col0,     kk, l16, half);
        v16bf bn = loadB_g(Wsrc, 2 * HDIM + col0, kk, l16, half);
        aR = wmma_bf(af, br, aR);
        aZ = wmma_bf(af, bz, aZ);
        if (!second) aGi = wmma_bf(af, bn, aGi);
        else         aGh = wmma_bf(af, bn, aGh);
      }
      int col = col0 + l16;
      #pragma unroll
      for (int e = 0; e < 8; ++e) {
        int row = m0 + half * 8 + e;
        float r = sigf(aR[e]);
        float z = sigf(aZ[e]);
        float n = tanhf(aGi[e] + r * aGh[e]);
        float ho = bf2f(h1old[row * HP + col]);
        h1new[row * HP + col] = f2bf((1.f - z) * n + z * ho);
      }
    }
    __syncthreads();

    // ---------- heads: y @ Wamp^T, y @ Wph^T (4 outputs / row) ----------
    {
      int row = t >> 2, o = t & 3;
      const float* Whead = (o < 2) ? (Wamp + o * HDIM) : (Wph + (o - 2) * HDIM);
      const uint4* yq = (const uint4*)h1new;
      float s = 0.f;
      #pragma unroll 4
      for (int qq = 0; qq < HDIM / 8; ++qq) {
        uint4 u = yq[row * (HP / 8) + qq];
        const float* wp = Whead + qq * 8;
        s += bflo(u.x) * wp[0] + bfhi(u.x) * wp[1]
           + bflo(u.y) * wp[2] + bfhi(u.y) * wp[3]
           + bflo(u.z) * wp[4] + bfhi(u.z) * wp[5]
           + bflo(u.w) * wp[6] + bfhi(u.w) * wp[7];
      }
      headsum[row * 4 + o] = s;
    }
    __syncthreads();

    // ---------- per-row softmax/softsign, symmetry mask, accumulate ----------
    if (t < BT) {
      int g = rowg0 + t;
      float a0 = headsum[t * 4 + 0] + bamp[0];
      float a1 = headsum[t * 4 + 1] + bamp[1];
      float mx = fmaxf(a0, a1);
      float e0 = __expf(a0 - mx), e1 = __expf(a1 - mx);
      float inv = 1.f / (e0 + e1);
      float ya0 = sqrtf(e0 * inv), ya1 = sqrtf(e1 * inv);
      float p0 = headsum[t * 4 + 2] + bph[0];
      float p1 = headsum[t * 4 + 3] + bph[1];
      const float PI = 3.14159265358979323846f;
      float yp0 = PI * p0 / (1.f + fabsf(p0));
      float yp1 = PI * p1 / (1.f + fabsf(p1));
      if (i >= 32) {                       // MIN_I
        float cnt   = ((i & 1) == 0) ? nupA[t] : ndnA[t];
        float lower = (float)((i >> 1) - 16);   // ALPHA - SORB/2 + i/2
        float occ_ok = (cnt < 16.f) ? 1.f : 0.f;
        float uno_ok = (cnt > lower) ? 1.f : 0.f;
        float m0v = ya0 * uno_ok, m1v = ya1 * occ_ok;
        float nrm = sqrtf(m0v * m0v + m1v * m1v + 1e-12f);
        ya0 = m0v / nrm; ya1 = m1v / nrm;
      }
      int b = xg[g * SORB + i];
      ampL[t] *= b ? ya1 : ya0;
      phL[t]  += b ? yp1 : yp0;
      if ((i & 1) == 0) nupA[t] += (float)b; else ndnA[t] += (float)b;
      bprev[t] = b;
    }
    __syncthreads();
    p ^= 1;
  }

  if (t < BT) {
    int g = rowg0 + t;
    out[2 * g]     = ampL[t];
    out[2 * g + 1] = phL[t];
  }
}

extern "C" void kernel_launch(void* const* d_in, const int* in_sizes, int n_in,
                              void* d_out, int out_size, void* d_ws, size_t ws_size,
                              hipStream_t stream) {
  const int*   xg    = (const int*)d_in[0];
  const float* Wih0  = (const float*)d_in[1];
  const float* Whh0f = (const float*)d_in[2];
  const float* Wih1f = (const float*)d_in[3];
  const float* Whh1f = (const float*)d_in[4];
  const float* Wamp  = (const float*)d_in[5];
  const float* bamp  = (const float*)d_in[6];
  const float* Wph   = (const float*)d_in[7];
  const float* bph   = (const float*)d_in[8];
  float* out = (float*)d_out;
  unsigned short* wbf = (unsigned short*)d_ws;   // 3 * WELEM bf16 = 4.5 MB

  int cvt_blocks = (3 * WELEM + 255) / 256;
  rnnwf_cvt<<<cvt_blocks, 256, 0, stream>>>(Whh0f, Wih1f, Whh1f, wbf);

  // dynamic LDS: 2 layers x 2 buffers x BT*HP bf16 + head/state scratch
  size_t shbytes = (size_t)4 * HBUF * 2 + 256 * 4 + 4 * BT * 4 + BT * 4;
  rnnwf_main<<<4096 / BT, 256, shbytes, stream>>>(
      xg, Wih0, wbf, wbf + WELEM, wbf + 2 * WELEM,
      Wamp, bamp, Wph, bph, out);
}
